// MultiHeadAttention_22969485099361
// MI455X (gfx1250) — matrix operand order repaired and checked by reading, not compiled, not run on hardware
//
#include <hip/hip_runtime.h>

typedef __attribute__((ext_vector_type(16))) __bf16         bf16x16;
typedef __attribute__((ext_vector_type(8)))  float          f32x8;
typedef __attribute__((ext_vector_type(4)))  unsigned int   u32x4;
typedef __attribute__((ext_vector_type(8)))  unsigned int   u32x8;
typedef __attribute__((ext_vector_type(4)))  unsigned short u16x4;
typedef __attribute__((ext_vector_type(8)))  int            i32x8;
typedef __attribute__((ext_vector_type(4)))  int            i32x4;

#define DEV __device__ __forceinline__

#if defined(__AMDGCN__) && __has_builtin(__builtin_amdgcn_tensor_load_to_lds)
#define HAVE_TDM 1
#else
#define HAVE_TDM 0
#endif

DEV unsigned short f2bf(float x) {
  unsigned u = __builtin_bit_cast(unsigned, x);
  u += 0x7FFFu + ((u >> 16) & 1u);          // round-to-nearest-even
  return (unsigned short)(u >> 16);
}

DEV bf16x16 frag_join(const unsigned short* p0, const unsigned short* p1) {
  u32x4 lo = *(const u32x4*)p0;
  u32x4 hi = *(const u32x4*)p1;
  u32x8 w  = __builtin_shufflevector(lo, hi, 0, 1, 2, 3, 4, 5, 6, 7);
  return __builtin_bit_cast(bf16x16, w);
}

// A-fragment (16x32 bf16): lane m = lane&15, half = lane>>4;
// elements e: k = (e<8 ? 0 : 16) + half*8 + (e&7)  -> two contiguous 16B chunks.
DEV bf16x16 load_a(const unsigned short* base, int row, int half, int st) {
  const unsigned short* p = base + (size_t)row * st;
  return frag_join(p + half * 8, p + 16 + half * 8);
}
// B-fragment (32x16 bf16) from K-contiguous buffer Bt[n][k]:
// lane n = lane&15, half = lane>>4; k = (e<8 ? 0 : 16) + half*8 + (e&7) -> two 16B chunks.
DEV bf16x16 load_b(const unsigned short* base, int n, int half, int st) {
  const unsigned short* p = base + (size_t)n * st;
  return frag_join(p + half * 8, p + 16 + half * 8);
}

DEV f32x8 wmma_bf16(bf16x16 a, bf16x16 b, f32x8 c) {
  return __builtin_amdgcn_wmma_f32_16x16x32_bf16(false, a, false, b, (short)0, c,
                                                 false, false);
}

#if HAVE_TDM
// Hardware LDS address = generic addr[31:0] (ISA aperture rule).
DEV unsigned lds_off(const void* p) { return (unsigned)(size_t)p; }

// 2D TDM load: tile_h rows x tile_w bf16 elements, row stride `stride` elems,
// LDS rows padded: (pad_amount+1) dwords after every 2^(pad_interval+1) dwords.
DEV void tdm_load_2d(unsigned ldsoff, const void* g, unsigned tw, unsigned th,
                     unsigned stride, unsigned pint, unsigned pamt) {
  unsigned long long ga = (unsigned long long)g;
  u32x4 g0;
  g0[0] = 1u;                                         // count=1 (user D#)
  g0[1] = ldsoff;                                     // lds_addr
  g0[2] = (unsigned)ga;                               // global_addr[31:0]
  g0[3] = (unsigned)((ga >> 32) & 0x01FFFFFFu) | (2u << 30);  // ga[56:32]|type=2
  i32x8 g1;
  g1[0] = (int)((1u << 16) | (1u << 20) | (pint << 22) | (pamt << 25));
  //            data_size=2B  pad_enable   pad_interval    pad_amount
  g1[1] = (int)((tw & 0xFFFFu) << 16);                          // tensor_dim0 lo
  g1[2] = (int)(((tw >> 16) & 0xFFFFu) | ((th & 0xFFFFu) << 16));  // td0 hi | td1 lo
  g1[3] = (int)(((th >> 16) & 0xFFFFu) | ((tw & 0xFFFFu) << 16));  // td1 hi | tile_dim0
  g1[4] = (int)(th & 0xFFFFu);                                  // tile_dim1
  g1[5] = (int)stride;                                          // tensor_dim0_stride
  g1[6] = 0;
  g1[7] = 0;
  i32x4 z4 = {0, 0, 0, 0};
  i32x8 z8 = {0, 0, 0, 0, 0, 0, 0, 0};
  __builtin_amdgcn_tensor_load_to_lds(g0, g1, z4, z4, z8, 0);
}

DEV void tdm_wait() {
#if __has_builtin(__builtin_amdgcn_s_wait_tensorcnt)
  __builtin_amdgcn_s_wait_tensorcnt(0);
#else
  asm volatile("s_wait_tensorcnt 0x0" ::: "memory");
#endif
}
#endif

// ---------------------------------------------------------------------------
// f32 -> bf16 bulk conversion (states)
// ---------------------------------------------------------------------------
__global__ void __launch_bounds__(256) cvt_kernel(const float* __restrict__ in,
                                                  unsigned short* __restrict__ out,
                                                  int n) {
  int i = (blockIdx.x * 256 + threadIdx.x) * 4;
  if (i + 3 < n) {
    const float4 v = *(const float4*)(in + i);
    u16x4 o;
    o.x = f2bf(v.x); o.y = f2bf(v.y); o.z = f2bf(v.z); o.w = f2bf(v.w);
    *(u16x4*)(out + i) = o;
  }
}

// ---------------------------------------------------------------------------
// f32 W[k][n] (2048x2048) -> bf16 Wt[n][k]  (transpose + convert, one-time)
// ---------------------------------------------------------------------------
__global__ void __launch_bounds__(256) cvtT_kernel(const float* __restrict__ in,
                                                   unsigned short* __restrict__ out) {
  __shared__ float t[32][33];
  const int bn = blockIdx.x * 32;
  const int bk = blockIdx.y * 32;
  const int lx = threadIdx.x & 31;
  const int ly = threadIdx.x >> 5;  // 0..7
#pragma unroll
  for (int i = 0; i < 32; i += 8)
    t[ly + i][lx] = in[(size_t)(bk + ly + i) * 2048 + bn + lx];
  __syncthreads();
#pragma unroll
  for (int i = 0; i < 32; i += 8)
    out[(size_t)(bn + ly + i) * 2048 + bk + lx] = f2bf(t[lx][ly + i]);
}

// ---------------------------------------------------------------------------
// bf16 GEMM: C[4096x2048] = A[4096x2048] @ Bt^T (Bt is [N][K], K-contiguous).
//   mode 0: f32 row-major out + bias
//   mode 1: bf16 [B,NH,S,HD] (Q/K), (acc+bias)*scale
//   mode 2: bf16 [B,NH,HD,S] (V transposed for attention)
// Block: 256 thr (8 waves), tile 128x128, BK=32, double-buffered TDM staging.
// ---------------------------------------------------------------------------
__global__ void __launch_bounds__(256) gemm_bf16_kernel(
    const unsigned short* __restrict__ A, const unsigned short* __restrict__ Bt,
    const float* __restrict__ bias, float scale,
    unsigned short* __restrict__ outb, float* __restrict__ outf, int mode) {
  constexpr int N = 2048, K = 2048, LD = 40;  // LD: 16 data + 4 pad dwords
  __shared__ unsigned short As[2][128 * LD];
  __shared__ unsigned short Bs[2][128 * LD];

  const int tid  = threadIdx.x;
  const int lane = tid & 31;
  const int wave = tid >> 5;
  const int nl   = lane & 15;
  const int half = lane >> 4;
  const int bm = blockIdx.y * 128;
  const int bn = blockIdx.x * 128;
  const int m0 = (wave >> 1) * 32;
  const int n0 = (wave & 1) * 64;

  const unsigned short* Abase = A + (size_t)bm * K;
  const unsigned short* Bbase = Bt + (size_t)bn * K;

  f32x8 acc[2][4] = {};

  auto stage = [&](int buf, int k0) {
#if HAVE_TDM
    if (tid < 32) {
      tdm_load_2d(lds_off(&As[buf][0]), Abase + k0, 32, 128, K, 3, 3);
      tdm_load_2d(lds_off(&Bs[buf][0]), Bbase + k0, 32, 128, K, 3, 3);
    }
#else
#pragma unroll
    for (int i = 0; i < 2; ++i) {
      int l = tid + i * 256;
      int row = l >> 2, kc = (l & 3) * 8;
      *(u32x4*)(&As[buf][row * LD + kc]) =
          *(const u32x4*)(Abase + (size_t)row * K + k0 + kc);
      *(u32x4*)(&Bs[buf][row * LD + kc]) =
          *(const u32x4*)(Bbase + (size_t)row * K + k0 + kc);
    }
#endif
  };
  auto stage_wait = [&]() {
#if HAVE_TDM
    if (tid < 32) tdm_wait();
#endif
  };

  int buf = 0;
  stage(0, 0);
  stage_wait();
  __syncthreads();

  for (int k0 = 0; k0 < K; k0 += 32) {
    if (k0 + 32 < K) stage(buf ^ 1, k0 + 32);  // DMA next tile during WMMA

    bf16x16 a0 = load_a(&As[buf][0], m0 + nl, half, LD);
    bf16x16 a1 = load_a(&As[buf][0], m0 + 16 + nl, half, LD);
#pragma unroll
    for (int j = 0; j < 4; ++j) {
      bf16x16 bf = load_b(&Bs[buf][0], n0 + j * 16 + nl, half, LD);
      acc[0][j] = wmma_bf16(a0, bf, acc[0][j]);
      acc[1][j] = wmma_bf16(a1, bf, acc[1][j]);
    }

    if (k0 + 32 < K) stage_wait();
    __syncthreads();
    buf ^= 1;
  }

  // Epilogue. C layout: lane n = nl, VGPR r -> m = r + 8*half.
#pragma unroll
  for (int mt = 0; mt < 2; ++mt) {
#pragma unroll
    for (int nt = 0; nt < 4; ++nt) {
      int gn = bn + n0 + nt * 16 + nl;
      float bv = bias[gn];
#pragma unroll
      for (int r = 0; r < 8; ++r) {
        int gm = bm + m0 + mt * 16 + r + 8 * half;
        float val = (acc[mt][nt][r] + bv) * scale;
        if (mode == 0) {
          outf[(size_t)gm * N + gn] = val;
        } else {
          int b = gm >> 11, s = gm & 2047;  // m = b*S + s
          int h = gn >> 7, d = gn & 127;    // n = h*HD + d
          if (mode == 1)
            outb[(((size_t)(b * 16 + h)) * 2048 + s) * 128 + d] = f2bf(val);
          else  // mode 2: [B,NH,HD,S]
            outb[(((size_t)(b * 16 + h)) * 128 + d) * 2048 + s] = f2bf(val);
        }
      }
    }
  }
}

// ---------------------------------------------------------------------------
// Flash attention (causal). Grid: (S/128, NH, B). Block: 256 thr (8 waves).
// K: [B,NH,S,HD]; V: [B,NH,HD,S] (pre-transposed). Double-buffered TDM staging.
// ---------------------------------------------------------------------------
__global__ void __launch_bounds__(256) attn_kernel(
    const unsigned short* __restrict__ Qp, const unsigned short* __restrict__ Kp,
    const unsigned short* __restrict__ Vt, unsigned short* __restrict__ Mg) {
  constexpr int S = 2048, HD = 128, NH = 16, F = 2048;
  constexpr int LDK = 136, LDV = 40, LDP = 40;
  __shared__ unsigned short Ks[2][32 * LDK];   // K chunk [kv][d]
  __shared__ unsigned short Vs[2][128 * LDV];  // V chunk [d][kv]
  __shared__ unsigned short Ps[8 * 16 * LDP];  // per-wave P tiles

  const int tid  = threadIdx.x;
  const int lane = tid & 31;
  const int wave = tid >> 5;
  const int nl   = lane & 15;
  const int half = lane >> 4;

  const int q0 = blockIdx.x * 128;
  const int hh = blockIdx.y;
  const int bb = blockIdx.z;
  const int qw = q0 + wave * 16;

  const size_t headoff = ((size_t)(bb * NH + hh)) * S * HD;
  const unsigned short* Q  = Qp + headoff;  // [S][HD]
  const unsigned short* Kh = Kp + headoff;  // [S][HD]
  const unsigned short* Vh = Vt + headoff;  // [HD][S]

  // Q A-fragments held in VGPRs for the whole KV loop (Q pre-scaled by 1/sqrt(HD)).
  bf16x16 qa[4];
#pragma unroll
  for (int ks = 0; ks < 4; ++ks)
    qa[ks] = load_a(Q + (size_t)qw * HD + ks * 32, nl, half, HD);

  f32x8 zacc[8] = {};
  float mrow[8], lrow[8];
#pragma unroll
  for (int r = 0; r < 8; ++r) { mrow[r] = -3.0e38f; lrow[r] = 0.f; }

  unsigned short* Pw = Ps + wave * 16 * LDP;

  auto stage = [&](int buf, int kv0) {
#if HAVE_TDM
    if (tid < 32) {
      tdm_load_2d(lds_off(&Ks[buf][0]), Kh + (size_t)kv0 * HD, 128, 32, HD, 5, 3);
      tdm_load_2d(lds_off(&Vs[buf][0]), Vh + kv0, 32, 128, S, 3, 3);
    }
#else
#pragma unroll
    for (int i = 0; i < 2; ++i) {
      int l = tid + i * 256;
      int k = l >> 4, dc = (l & 15) * 8;
      *(u32x4*)(&Ks[buf][k * LDK + dc]) =
          *(const u32x4*)(Kh + (size_t)(kv0 + k) * HD + dc);
      int row = l >> 2, kc = (l & 3) * 8;
      *(u32x4*)(&Vs[buf][row * LDV + kc]) =
          *(const u32x4*)(Vh + (size_t)row * S + kv0 + kc);
    }
#endif
  };
  auto stage_wait = [&]() {
#if HAVE_TDM
    if (tid < 32) tdm_wait();
#endif
  };

  const int kvend = q0 + 128;
  int buf = 0;
  stage(0, 0);
  stage_wait();
  __syncthreads();

  for (int kv0 = 0; kv0 < kvend; kv0 += 32) {
    if (kv0 + 32 < kvend) stage(buf ^ 1, kv0 + 32);

    // Scores: S[16x32] = Q[16x128] @ K^T (B-frag k index = d, contiguous in Ks rows).
    f32x8 s0 = {}, s1 = {};
#pragma unroll
    for (int ks = 0; ks < 4; ++ks) {
      bf16x16 b0 = load_b(&Ks[buf][0] + ks * 32, nl, half, LDK);
      bf16x16 b1 = load_b(&Ks[buf][0] + ks * 32, 16 + nl, half, LDK);
      s0 = wmma_bf16(qa[ks], b0, s0);
      s1 = wmma_bf16(qa[ks], b1, s1);
    }

    // Online softmax per row (= r + 8*half; columns span the 16-lane half-group).
#pragma unroll
    for (int r = 0; r < 8; ++r) {
      int qi = qw + r + 8 * half;
      float v0 = ((kv0 + nl) <= qi) ? s0[r] : -3.0e38f;
      float v1 = ((kv0 + 16 + nl) <= qi) ? s1[r] : -3.0e38f;
      float mx = fmaxf(v0, v1);
      mx = fmaxf(mx, __shfl_xor(mx, 1, 32));
      mx = fmaxf(mx, __shfl_xor(mx, 2, 32));
      mx = fmaxf(mx, __shfl_xor(mx, 4, 32));
      mx = fmaxf(mx, __shfl_xor(mx, 8, 32));
      float mnew = fmaxf(mrow[r], mx);
      float sc = __expf(mrow[r] - mnew);
      float p0 = __expf(v0 - mnew);
      float p1 = __expf(v1 - mnew);
      float sum = p0 + p1;
      sum += __shfl_xor(sum, 1, 32);
      sum += __shfl_xor(sum, 2, 32);
      sum += __shfl_xor(sum, 4, 32);
      sum += __shfl_xor(sum, 8, 32);
      lrow[r] = lrow[r] * sc + sum;
      mrow[r] = mnew;
#pragma unroll
      for (int t = 0; t < 8; ++t) zacc[t][r] *= sc;
      unsigned short* pr = Pw + (r + 8 * half) * LDP;
      pr[nl]      = f2bf(p0);
      pr[16 + nl] = f2bf(p1);
    }
    __syncthreads();  // Ps visibility (and orders LDS traffic)

    // Z[16x128] += P[16x32] @ V[32x128]
    bf16x16 pa = load_a(Pw, nl, half, LDP);
#pragma unroll
    for (int t = 0; t < 8; ++t) {
      bf16x16 vb = load_b(&Vs[buf][0], t * 16 + nl, half, LDV);
      zacc[t] = wmma_bf16(pa, vb, zacc[t]);
    }

    if (kv0 + 32 < kvend) stage_wait();
    __syncthreads();
    buf ^= 1;
  }

  // Normalize and write merged context [B,S,NH*HD] as bf16.
#pragma unroll
  for (int t = 0; t < 8; ++t) {
    int d = t * 16 + nl;
#pragma unroll
    for (int r = 0; r < 8; ++r) {
      int qi = qw + r + 8 * half;
      float val = zacc[t][r] / lrow[r];
      Mg[((size_t)(bb * S + qi)) * F + hh * HD + d] = f2bf(val);
    }
  }
}

// ---------------------------------------------------------------------------
extern "C" void kernel_launch(void* const* d_in, const int* in_sizes, int n_in,
                              void* d_out, int out_size, void* d_ws, size_t ws_size,
                              hipStream_t stream) {
  const float* states = (const float*)d_in[0];
  const float* Wq = (const float*)d_in[1];
  const float* bq = (const float*)d_in[2];
  const float* Wk = (const float*)d_in[3];
  const float* bk = (const float*)d_in[4];
  const float* Wv = (const float*)d_in[5];
  const float* bv = (const float*)d_in[6];
  const float* Wo = (const float*)d_in[7];
  const float* bo = (const float*)d_in[8];
  float* out = (float*)d_out;

  const size_t NSX = 8388608;  // B*S*F
  const size_t NW  = 4194304;  // F*F
  unsigned short* sx  = (unsigned short*)d_ws;
  unsigned short* wqT = sx + NSX;   // transposed bf16 weights [n][k]
  unsigned short* wkT = wqT + NW;
  unsigned short* wvT = wkT + NW;
  unsigned short* woT = wvT + NW;
  unsigned short* Qb  = woT + NW;   // [B,NH,S,HD] bf16
  unsigned short* Kb  = Qb + NSX;   // [B,NH,S,HD] bf16
  unsigned short* Vtr = Kb + NSX;   // [B,NH,HD,S] bf16
  unsigned short* Mg  = Vtr + NSX;  // merged context bf16

  cvt_kernel<<<NSX / 1024, 256, 0, stream>>>(states, sx, (int)NSX);
  dim3 gt(64, 64);
  cvtT_kernel<<<gt, 256, 0, stream>>>(Wq, wqT);
  cvtT_kernel<<<gt, 256, 0, stream>>>(Wk, wkT);
  cvtT_kernel<<<gt, 256, 0, stream>>>(Wv, wvT);
  cvtT_kernel<<<gt, 256, 0, stream>>>(Wo, woT);

  dim3 gg(16, 32);  // (N/128, M/128)
  gemm_bf16_kernel<<<gg, 256, 0, stream>>>(sx, wqT, bq, 0.08838834764831845f,
                                           Qb, nullptr, 1);
  gemm_bf16_kernel<<<gg, 256, 0, stream>>>(sx, wkT, bk, 1.0f, Kb, nullptr, 1);
  gemm_bf16_kernel<<<gg, 256, 0, stream>>>(sx, wvT, bv, 1.0f, Vtr, nullptr, 2);

  attn_kernel<<<dim3(16, 16, 2), 256, 0, stream>>>(Qb, Kb, Vtr, Mg);

  gemm_bf16_kernel<<<gg, 256, 0, stream>>>(Mg, woT, bo, 1.0f, nullptr, out, 0);
}
